// SemanticLSTM_19576460935338
// MI455X (gfx1250) — compile-verified, matrix operand order checked by reading, not run on hardware
//
#include <hip/hip_runtime.h>
#include <hip/hip_bf16.h>
#include <math.h>

// Problem constants (match reference)
#define BB 256
#define TT 80
#define EE 512
#define II 1024
#define HH 1024
#define LL 300
#define FF 2048

typedef __attribute__((ext_vector_type(16))) __bf16 v16bf;
typedef __attribute__((ext_vector_type(8)))  float  v8f;

union FragBF {
    uint4 q[2];   // 32 bytes
    v16bf v;      // 16 x bf16 = 32 bytes
};

__device__ __forceinline__ unsigned short f2bf(float f) {
    unsigned int u = __float_as_uint(f);
    u += 0x7FFFu + ((u >> 16) & 1u);   // round-to-nearest-even
    return (unsigned short)(u >> 16);
}

// ---------------------------------------------------------------------------
// Repack a (4, K, N) fp32 weight into bf16 WMMA B-fragment order:
// P[ ((g*Nt + nt)*Kt + kt)*32 + lane ][ 0..15 ]  (16 contiguous bf16 per lane)
// Element e of lane: half=lane>>4, n=nt*16+(lane&15),
//   k = kt*32 + (e<8 ? half*8+e : 16+half*8+(e-8))     (ISA 7.12.2, 16-bit B)
// ---------------------------------------------------------------------------
__global__ __launch_bounds__(256)
void repack_bf16_kernel(const float* __restrict__ W, unsigned short* __restrict__ P,
                        int K, int N) {
    int Kt = K >> 5, Nt = N >> 4;
    int idx = blockIdx.x * blockDim.x + threadIdx.x;
    int total = 4 * Nt * Kt * 32;
    if (idx >= total) return;
    int lane = idx & 31;
    int t = idx >> 5;
    int kt = t % Kt; t /= Kt;
    int nt = t % Nt; int g = t / Nt;
    int half = lane >> 4, nl = lane & 15;
    int n = nt * 16 + nl;
    unsigned short* dst = P + (size_t)idx * 16;
#pragma unroll
    for (int e = 0; e < 16; ++e) {
        int kl = (e < 8) ? (half * 8 + e) : (16 + half * 8 + (e - 8));
        int k = kt * 32 + kl;
        dst[e] = f2bf(W[((size_t)g * K + k) * N + n]);
    }
}

// ---------------------------------------------------------------------------
// One-time projections: XP = (emb@Wa)*(sem@Wb), VP = (cnn@Ca)*(sem@Cb),
// SH = sem@Ub.  Layout [ (b*4+g)*I + i ], i fastest -> coalesced weight reads.
// ---------------------------------------------------------------------------
__global__ __launch_bounds__(256)
void precompute_proj_kernel(const int* __restrict__ cap,
                            const float* __restrict__ embed,
                            const float* __restrict__ sem,
                            const float* __restrict__ cnn,
                            const float* __restrict__ Wa, const float* __restrict__ Wb,
                            const float* __restrict__ Ub,
                            const float* __restrict__ Ca, const float* __restrict__ Cb,
                            float* __restrict__ XP, float* __restrict__ VP,
                            float* __restrict__ SH) {
    int idx = blockIdx.x * blockDim.x + threadIdx.x;
    if (idx >= BB * 4 * II) return;
    int i = idx & (II - 1);
    int g = (idx >> 10) & 3;
    int b = idx >> 12;
    int tok = cap[b * TT];                       // captions[:,0]
    const float* ev = embed + (size_t)tok * EE;
    const float* sv = sem + (size_t)b * LL;
    const float* cv = cnn + (size_t)b * FF;

    float x1 = 0.f;
    for (int e = 0; e < EE; ++e) x1 += ev[e] * Wa[((size_t)g * EE + e) * II + i];
    float x2 = 0.f, sh = 0.f, v2 = 0.f;
    for (int l = 0; l < LL; ++l) {
        float s = sv[l];
        x2 += s * Wb[((size_t)g * LL + l) * II + i];
        sh += s * Ub[((size_t)g * LL + l) * II + i];
        v2 += s * Cb[((size_t)g * LL + l) * II + i];
    }
    float v1 = 0.f;
    for (int f = 0; f < FF; ++f) v1 += cv[f] * Ca[((size_t)g * FF + f) * II + i];

    XP[idx] = x1 * x2;
    VP[idx] = v1 * v2;
    SH[idx] = sh;
}

// base = XP@Wc + VP@Cc + bias   -> [ (b*4+g)*H + h ]
__global__ __launch_bounds__(256)
void precompute_base_kernel(const float* __restrict__ XP, const float* __restrict__ VP,
                            const float* __restrict__ Wc, const float* __restrict__ Cc,
                            const float* __restrict__ bias, float* __restrict__ base) {
    int idx = blockIdx.x * blockDim.x + threadIdx.x;
    if (idx >= BB * 4 * HH) return;
    int h = idx & (HH - 1);
    int g = (idx >> 10) & 3;
    int b = idx >> 12;
    const float* xp = XP + ((size_t)b * 4 + g) * II;
    const float* vp = VP + ((size_t)b * 4 + g) * II;
    float acc = bias[g * HH + h];
    for (int i = 0; i < II; ++i) {
        acc += xp[i] * Wc[((size_t)g * II + i) * HH + h];
        acc += vp[i] * Cc[((size_t)g * II + i) * HH + h];
    }
    base[idx] = acc;
}

// zero h (bf16) and c (f32)
__global__ __launch_bounds__(256)
void init_state_kernel(unsigned short* __restrict__ hb, float* __restrict__ c) {
    int idx = blockIdx.x * blockDim.x + threadIdx.x;
    if (idx >= BB * HH) return;
    hb[idx] = 0;
    c[idx] = 0.f;
}

// ---------------------------------------------------------------------------
// Step GEMM 1: R[b,g,i] = bf16( (h @ Ua[g])[b,i] * SH[b,g,i] )
// Block = 8 waves sharing one 32-row A panel staged in LDS (66 KB, padded
// +16B/row so the 16 lanes of a half-wave hit 16 distinct bank groups on
// ds_load_b128).  Each wave: 32 rows x 16 cols x 4 gates = 8 accumulators,
// 8 WMMAs per k-tile against 4 ds loads + 8 global b128 loads.
// ---------------------------------------------------------------------------
__global__ __launch_bounds__(256)
void step_gemm1_kernel(const unsigned short* __restrict__ hb,
                       const unsigned short* __restrict__ UaP,
                       const float* __restrict__ SH,
                       unsigned short* __restrict__ R) {
    const int NT = II / 16, KT = HH / 32;
    __shared__ unsigned short As[32][1032];   // 32 x 1024 data + 8 pad
    int mb = blockIdx.x >> 3;                 // 0..7  (32-row panel)
    int ng = blockIdx.x & 7;                  // 0..7  (group of 8 n-tiles)

    // cooperative stage of the A panel (gate- and wave-invariant)
    for (int j = threadIdx.x; j < 32 * 128; j += 256) {
        int row = j >> 7, c8 = (j & 127) * 8;
        *(uint4*)&As[row][c8] =
            *(const uint4*)(hb + (size_t)(mb * 32 + row) * HH + c8);
    }
    __syncthreads();

    int w = threadIdx.x >> 5, lane = threadIdx.x & 31;
    int nt = ng * 8 + w;
    int half = lane >> 4, ln = lane & 15;

    v8f acc[2][4] = {};
    for (int kt = 0; kt < KT; ++kt) {
        int k0 = kt * 32;
        FragBF a[2], b[4];
#pragma unroll
        for (int s = 0; s < 2; ++s) {
            a[s].q[0] = *(const uint4*)&As[s * 16 + ln][k0 + half * 8];
            a[s].q[1] = *(const uint4*)&As[s * 16 + ln][k0 + 16 + half * 8];
        }
#pragma unroll
        for (int g = 0; g < 4; ++g) {
            const unsigned short* bp =
                UaP + ((size_t)(g * NT + nt) * KT + kt) * 512 + lane * 16;
            b[g].q[0] = *(const uint4*)bp;
            b[g].q[1] = *(const uint4*)(bp + 8);
            if (kt + 1 < KT)
                __builtin_prefetch(bp + 512, 0, 3);   // next k-tile, WGP scope
        }
#pragma unroll
        for (int g = 0; g < 4; ++g)
#pragma unroll
            for (int s = 0; s < 2; ++s)
                acc[s][g] = __builtin_amdgcn_wmma_f32_16x16x32_bf16(
                    false, a[s].v, false, b[g].v, (short)0, acc[s][g], false, false);
    }

    // D layout: VGPR r -> M = r + 8*half, N = ln  (ISA 7.12.2)
    int n = nt * 16 + ln;
#pragma unroll
    for (int s = 0; s < 2; ++s)
#pragma unroll
        for (int g = 0; g < 4; ++g)
#pragma unroll
            for (int r = 0; r < 8; ++r) {
                int brow = mb * 32 + s * 16 + r + 8 * half;
                size_t o = ((size_t)brow * 4 + g) * II + n;
                R[o] = f2bf(acc[s][g][r] * SH[o]);
            }
}

// ---------------------------------------------------------------------------
// Step GEMM 2 + fused LSTM cell: one wave computes the 16x16 (b,h) tile for
// ALL 4 gates (4 accumulators, 4 WMMAs/k-tile), so the epilogue has the full
// pre-activation quad per element and applies the cell update in-register:
//   c = sig(p1)*c + sig(p0)*tanh(p3) ; h = sig(p2)*tanh(c)
// Writes c (f32), h (f32 -> d_out), h (bf16 -> next step's A).
// ---------------------------------------------------------------------------
__global__ __launch_bounds__(256)
void step_gemm2_cell_kernel(const unsigned short* __restrict__ R,
                            const unsigned short* __restrict__ UcP,
                            const float* __restrict__ base,
                            float* __restrict__ c,
                            unsigned short* __restrict__ hb_out,
                            float* __restrict__ hout) {
    const int NT = HH / 16, KT = II / 32;
    int wid = blockIdx.x * (blockDim.x >> 5) + (threadIdx.x >> 5);
    int lane = threadIdx.x & 31;
    int nt = wid % NT;
    int mt = wid / NT;                 // 0..15
    int half = lane >> 4, ln = lane & 15;
    int m = mt * 16 + ln;

    v8f acc[4] = {};
    for (int kt = 0; kt < KT; ++kt) {
        int k0 = kt * 32;
#pragma unroll
        for (int g = 0; g < 4; ++g) {
            FragBF a, b;
            const unsigned short* ar = R + ((size_t)m * 4 + g) * II;
            a.q[0] = *(const uint4*)(ar + k0 + half * 8);
            a.q[1] = *(const uint4*)(ar + k0 + 16 + half * 8);
            const unsigned short* bp =
                UcP + ((size_t)(g * NT + nt) * KT + kt) * 512 + lane * 16;
            b.q[0] = *(const uint4*)bp;
            b.q[1] = *(const uint4*)(bp + 8);
            if (kt + 1 < KT)
                __builtin_prefetch(bp + 512, 0, 3);
            acc[g] = __builtin_amdgcn_wmma_f32_16x16x32_bf16(
                false, a.v, false, b.v, (short)0, acc[g], false, false);
        }
    }

    int n = nt * 16 + ln;
#pragma unroll
    for (int r = 0; r < 8; ++r) {
        int brow = mt * 16 + r + 8 * half;
        size_t ob = ((size_t)brow * 4) * HH + n;
        float p0 = acc[0][r] + base[ob];
        float p1 = acc[1][r] + base[ob + HH];
        float p2 = acc[2][r] + base[ob + 2 * HH];
        float p3 = acc[3][r] + base[ob + 3 * HH];
        float ig = 1.f / (1.f + __expf(-p0));
        float fg = 1.f / (1.f + __expf(-p1));
        float og = 1.f / (1.f + __expf(-p2));
        float gg = tanhf(p3);
        size_t oh = (size_t)brow * HH + n;
        float cn = fg * c[oh] + ig * gg;
        c[oh] = cn;
        float hn = og * tanhf(cn);
        hout[oh] = hn;
        hb_out[oh] = f2bf(hn);
    }
}

// ---------------------------------------------------------------------------
extern "C" void kernel_launch(void* const* d_in, const int* in_sizes, int n_in,
                              void* d_out, int out_size, void* d_ws, size_t ws_size,
                              hipStream_t stream) {
    const int*   cap   = (const int*)d_in[0];
    const float* cnn   = (const float*)d_in[1];
    const float* sem   = (const float*)d_in[2];
    const float* embed = (const float*)d_in[3];
    const float* Wa    = (const float*)d_in[4];
    const float* Wb    = (const float*)d_in[5];
    const float* Wc    = (const float*)d_in[6];
    const float* Ua    = (const float*)d_in[7];
    const float* Ub    = (const float*)d_in[8];
    const float* Uc    = (const float*)d_in[9];
    const float* Ca    = (const float*)d_in[10];
    const float* Cb    = (const float*)d_in[11];
    const float* Cc    = (const float*)d_in[12];
    const float* bias  = (const float*)d_in[13];
    float* hout = (float*)d_out;

    // workspace layout
    char* ws = (char*)d_ws;
    unsigned short* UaP = (unsigned short*)ws; ws += (size_t)4 * HH * II * 2;
    unsigned short* UcP = (unsigned short*)ws; ws += (size_t)4 * II * HH * 2;
    float* SHb = (float*)ws; ws += (size_t)BB * 4 * II * 4;
    float* XP  = (float*)ws; ws += (size_t)BB * 4 * II * 4;
    float* VP  = (float*)ws; ws += (size_t)BB * 4 * II * 4;
    float* baseB = (float*)ws; ws += (size_t)BB * 4 * HH * 4;
    unsigned short* R  = (unsigned short*)ws; ws += (size_t)BB * 4 * II * 2;
    unsigned short* hb = (unsigned short*)ws; ws += (size_t)BB * HH * 2;
    float* cst = (float*)ws; ws += (size_t)BB * HH * 4;

    const int TPB = 256;

    // 1. repack recurrent weights to bf16 fragment order (amortized over 80 steps)
    {
        int total = 4 * (II / 16) * (HH / 32) * 32;
        repack_bf16_kernel<<<(total + TPB - 1) / TPB, TPB, 0, stream>>>(Ua, UaP, HH, II);
        total = 4 * (HH / 16) * (II / 32) * 32;
        repack_bf16_kernel<<<(total + TPB - 1) / TPB, TPB, 0, stream>>>(Uc, UcP, II, HH);
    }

    // 2. one-time projections and base
    {
        int total = BB * 4 * II;
        precompute_proj_kernel<<<(total + TPB - 1) / TPB, TPB, 0, stream>>>(
            cap, embed, sem, cnn, Wa, Wb, Ub, Ca, Cb, XP, VP, SHb);
        total = BB * 4 * HH;
        precompute_base_kernel<<<(total + TPB - 1) / TPB, TPB, 0, stream>>>(
            XP, VP, Wc, Cc, bias, baseB);
    }

    // 3. init recurrent state
    init_state_kernel<<<(BB * HH + TPB - 1) / TPB, TPB, 0, stream>>>(hb, cst);

    // 4. T=80 recurrent steps (serial dependency -> 2 launches per step)
    int g1Blocks = (BB / 32) * ((II / 16) / 8);           // 8 * 8 = 64 blocks
    int g2Blocks = ((BB / 16) * (HH / 16)) / (TPB / 32);  // 1024 waves -> 128 blocks
    for (int t = 0; t < TT; ++t) {
        step_gemm1_kernel<<<g1Blocks, TPB, 0, stream>>>(hb, UaP, SHb, R);
        step_gemm2_cell_kernel<<<g2Blocks, TPB, 0, stream>>>(R, UcP, baseB, cst, hb, hout);
    }
}
